// Res_Block_Down_3D_Sparse_84499186581675
// MI455X (gfx1250) — matrix operand order repaired and checked by reading, not compiled, not run on hardware
//
#include <hip/hip_runtime.h>
#include <hip/hip_bf16.h>

// ---------------------------------------------------------------------------
// Sparse 3D residual block (SubmanifoldConv -> maskedBN -> ReLU ->
// SubmanifoldConv -> maskedBN, + NetworkInNetwork residual, 2x2x2 avg pool),
// dense-with-masking, matmuls on CDNA5 WMMA bf16, weights staged to LDS with
// the Tensor Data Mover (tensor_load_to_lds + s_wait_tensorcnt).
// ---------------------------------------------------------------------------

typedef __attribute__((ext_vector_type(16))) __bf16 v16bf;
typedef __attribute__((ext_vector_type(8)))  __bf16 v8bf;
typedef __attribute__((ext_vector_type(8)))  float  v8f;
typedef __attribute__((ext_vector_type(4)))  unsigned int u32x4;
typedef __attribute__((ext_vector_type(8)))  int i32x8;
typedef __attribute__((ext_vector_type(4)))  int i32x4;

static constexpr int   BNUM  = 2;
static constexpr int   DGRID = 64;
static constexpr int   CIN   = 32;
static constexpr int   COUT  = 64;
static constexpr int   NSITE = BNUM * DGRID * DGRID * DGRID;   // 524288
static constexpr float EPSV  = 1e-4f;

// ---------------------------------------------------------------------------
// TDM: 1-D bulk copy global -> LDS via Tensor DMA descriptor (ISA ch.8).
// data_size = 4B; tensor/tile dim0 = ndwords (< 65536).
// ---------------------------------------------------------------------------
__device__ __forceinline__ unsigned lds_off_of(const void* p) {
    // flat shared address: low 32 bits are the LDS byte offset (ISA 10.2)
    return (unsigned)(unsigned long long)(uintptr_t)p;
}

__device__ __forceinline__ void tdm_load_to_lds(unsigned lds_off, const void* gsrc,
                                                unsigned ndwords) {
#if __has_builtin(__builtin_amdgcn_tensor_load_to_lds)
    unsigned long long ga = (unsigned long long)(uintptr_t)gsrc;
    u32x4 g0 = {0u, 0u, 0u, 0u};
    g0[0] = 1u;                                        // count=1, user mode
    g0[1] = lds_off;                                   // lds_addr (bytes)
    g0[2] = (unsigned)(ga & 0xFFFFFFFFu);              // global_addr[31:0]
    g0[3] = (unsigned)((ga >> 32) & 0x1FFFFFFu)        // global_addr[56:32]
          | (2u << 30);                                // type = 2 ("image")
    i32x8 g1 = {0, 0, 0, 0, 0, 0, 0, 0};
    g1[0] = (int)(2u << 16);                           // data_size = 4 bytes
    g1[1] = (int)((ndwords & 0xFFFFu) << 16);          // tensor_dim0[15:0]  @ bits 63:48
    g1[2] = (int)(((ndwords >> 16) & 0xFFFFu)          // tensor_dim0[31:16] @ bits 79:64
          | (1u << 16));                               // tensor_dim1 = 1    @ bits 111:80
    g1[3] = (int)((ndwords & 0xFFFFu) << 16);          // tile_dim0 @ bits 127:112
    g1[5] = (int)ndwords;                              // tensor_dim0_stride low32
    i32x4 z4 = {0, 0, 0, 0};
#if defined(__clang_major__) && (__clang_major__ >= 23)
    i32x8 z8 = {0, 0, 0, 0, 0, 0, 0, 0};
    __builtin_amdgcn_tensor_load_to_lds(g0, g1, z4, z4, z8, 0);
#else
    __builtin_amdgcn_tensor_load_to_lds(g0, g1, z4, z4, 0);
#endif
    __builtin_amdgcn_s_wait_tensorcnt(0);
#else
    (void)lds_off; (void)gsrc; (void)ndwords;
#endif
}

// ---------------------------------------------------------------------------
// helpers
// ---------------------------------------------------------------------------
__global__ void k_zero_f32(float* __restrict__ p, int n) {
    int i = blockIdx.x * blockDim.x + threadIdx.x;
    if (i < n) p[i] = 0.0f;
}

// x = feats * mask (bf16), cnt = sum(mask)
__global__ void k_mask_convert(const float* __restrict__ feats,
                               const unsigned char* __restrict__ mask,
                               __bf16* __restrict__ xb,
                               float* __restrict__ cnt) {
    int s = blockIdx.x * blockDim.x + threadIdx.x;
    __shared__ float scnt;
    if (threadIdx.x == 0) scnt = 0.0f;
    __syncthreads();
    float mf = 0.0f;
    if (s < NSITE) {
        mf = mask[s] ? 1.0f : 0.0f;
        const float4* fp = (const float4*)(feats + (size_t)s * CIN);
        __bf16*       xp = xb + (size_t)s * CIN;
#pragma unroll
        for (int c = 0; c < CIN / 8; ++c) {
            float4 a = fp[2 * c], b = fp[2 * c + 1];
            v8bf o;
            o[0] = (__bf16)(a.x * mf); o[1] = (__bf16)(a.y * mf);
            o[2] = (__bf16)(a.z * mf); o[3] = (__bf16)(a.w * mf);
            o[4] = (__bf16)(b.x * mf); o[5] = (__bf16)(b.y * mf);
            o[6] = (__bf16)(b.z * mf); o[7] = (__bf16)(b.w * mf);
            *(v8bf*)(xp + 8 * c) = o;
        }
    }
    atomicAdd(&scnt, mf);
    __syncthreads();
    if (threadIdx.x == 0) atomicAdd(cnt, scnt);
}

// ---------------------------------------------------------------------------
// Prepack f32 weights [TAPS][KIN][64] into per-lane WMMA B fragments, bf16:
//   frag[((t*KSTEPS+kstep)*4 + nTile)*32 + lane][16]
// B layout: lanes 0-15 hold K 0..7,16..23 of the 32-K chunk, lanes 16-31 K 8..15,24..31;
// column n = (lane&15) + nTile*16; element idx<8 -> K khalf*8+idx, idx>=8 -> K 16+khalf*8+idx-8.
// ---------------------------------------------------------------------------
template <int KIN, int TAPS>
__global__ void k_prepack(const float* __restrict__ w, __bf16* __restrict__ frag) {
    constexpr int KSTEPS = KIN / 32;
    constexpr int TOT    = TAPS * KSTEPS * 2048;       // 4 nTile * 32 lane * 16 idx
    int e = blockIdx.x * blockDim.x + threadIdx.x;
    if (e >= TOT) return;
    const int idx   = e & 15;
    const int lane  = (e >> 4) & 31;
    const int nTile = (e >> 9) & 3;
    const int group = e >> 11;                         // t*KSTEPS + kstep
    const int t     = group / KSTEPS;
    const int kstep = group % KSTEPS;
    const int khalf = lane >> 4;
    const int n     = (lane & 15) + nTile * 16;
    const int k     = kstep * 32 + khalf * 8 + (idx & 7) + (idx >> 3) * 16;
    frag[e] = (__bf16)w[((size_t)t * KIN + k) * 64 + n];
}

// ---------------------------------------------------------------------------
// Implicit-GEMM 3x3x3 conv on v_wmma_f32_16x16x32_bf16.
// Block = 256 threads (8 wave32) = 32 sites x 64 out channels;
// wave tile = 16 sites x 16 channels.  All B fragments staged in LDS via TDM.
// Fused epilogue: +bias, *mask, f32 store, per-channel sum/sumsq for BN.
// ---------------------------------------------------------------------------
template <int KIN>
__global__ void k_conv3(const __bf16* __restrict__ xin,    // [NSITE][KIN]
                        const __bf16* __restrict__ wfrag,  // prepacked fragments
                        const float*  __restrict__ bias,
                        const unsigned char* __restrict__ mask,
                        float* __restrict__ hout,          // [NSITE][64]
                        float* __restrict__ gsum, float* __restrict__ gsq) {
    constexpr int      KSTEPS = KIN / 32;
    constexpr unsigned WBYTES = 27u * KSTEPS * 4 * 32 * 16 * 2;

    extern __shared__ __align__(16) char smem_raw[];
    __bf16* swt = (__bf16*)smem_raw;

    // ---- stage all weight fragments into LDS (one TDM op from wave 0) ----
#if __has_builtin(__builtin_amdgcn_tensor_load_to_lds)
    if (threadIdx.x < 32)
        tdm_load_to_lds(lds_off_of(swt), wfrag, WBYTES / 4);
#else
    for (unsigned i = threadIdx.x * 8u; i * 2u < WBYTES; i += 256u * 8u)
        *(v8bf*)(swt + i) = *(const v8bf*)(wfrag + i);
#endif
    __syncthreads();

    const int lane  = threadIdx.x & 31;
    const int wave  = threadIdx.x >> 5;
    const int mTile = wave >> 2;          // 0..1
    const int nTile = wave & 3;           // 0..3
    const int m     = lane & 15;
    const int khalf = lane >> 4;
    const int siteBase = blockIdx.x * 32 + mTile * 16;
    const int s  = siteBase + m;
    const int wx = s & 63, hy = (s >> 6) & 63, dz = (s >> 12) & 63, bb = s >> 18;
    const int n  = (lane & 15) + nTile * 16;

    const v16bf* sfrag = (const v16bf*)swt;
    v8f acc = {};

    int t = 0;
    for (int ddz = -1; ddz <= 1; ++ddz)
    for (int ddy = -1; ddy <= 1; ++ddy)
    for (int ddx = -1; ddx <= 1; ++ddx, ++t) {
        const int nd = dz + ddz, nh = hy + ddy, nw = wx + ddx;
        const bool rowok = ((unsigned)nd < 64u) && ((unsigned)nh < 64u);
#pragma unroll
        for (int kstep = 0; kstep < KSTEPS; ++kstep) {
            const int kb = kstep * 32;
            // A fragment (ISA 16-bit A 16x32 layout): two b128 loads per lane
            v8bf alo = {}, ahi = {};
            if (rowok && ((unsigned)nw < 64u)) {
                const __bf16* ap = xin +
                    ((((size_t)bb * 64 + nd) * 64 + nh) * 64 + nw) * KIN + kb + khalf * 8;
                alo = *(const v8bf*)ap;          // K = kb + khalf*8 .. +7
                ahi = *(const v8bf*)(ap + 16);   // K = kb + 16 + khalf*8 .. +7
            }
            union { v16bf v; v8bf h[2]; } A;
            A.h[0] = alo; A.h[1] = ahi;

            // B fragment: one 32B LDS read (ds_load_b128 x2)
            const v16bf Bv = sfrag[((t * KSTEPS + kstep) * 4 + nTile) * 32 + lane];

            acc = __builtin_amdgcn_wmma_f32_16x16x32_bf16(
                false, A.v, false, Bv, (short)0, acc, false, false);
        }
    }

    // ---- epilogue: bias, mask, store, per-channel BN statistics ----
    __shared__ float ssum[64];
    __shared__ float ssq[64];
    if (threadIdx.x < 64) { ssum[threadIdx.x] = 0.0f; ssq[threadIdx.x] = 0.0f; }
    __syncthreads();

    const float bv = bias[n];
    float ps = 0.0f, pq = 0.0f;
#pragma unroll
    for (int r = 0; r < 8; ++r) {
        const int mo = r + khalf * 8;              // C/D: lanes>=16 hold M 8..15
        const int so = siteBase + mo;
        const float mf = mask[so] ? 1.0f : 0.0f;
        const float v = (acc[r] + bv) * mf;
        hout[(size_t)so * 64 + n] = v;
        ps += v; pq += v * v;
    }
    atomicAdd(&ssum[n], ps);
    atomicAdd(&ssq[n],  pq);
    __syncthreads();
    if (threadIdx.x < 64) {
        atomicAdd(&gsum[threadIdx.x], ssum[threadIdx.x]);
        atomicAdd(&gsq[threadIdx.x],  ssq[threadIdx.x]);
    }
}

// scale = g*rsqrt(var+eps), shift = be - mean*scale
__global__ void k_bn_finalize(const float* __restrict__ cnt,
                              const float* __restrict__ sum,
                              const float* __restrict__ sq,
                              const float* __restrict__ g,
                              const float* __restrict__ be,
                              float* __restrict__ scale,
                              float* __restrict__ shift) {
    int nidx = threadIdx.x;
    if (nidx < 64) {
        float c    = fmaxf(cnt[0], 1.0f);
        float mean = sum[nidx] / c;
        float var  = sq[nidx] / c - mean * mean;
        float rs   = rsqrtf(var + EPSV);
        float sc   = g[nidx] * rs;
        scale[nidx] = sc;
        shift[nidx] = be[nidx] - mean * sc;
    }
}

// h1b = bf16( relu( (h*scale + shift) * mask ) )
__global__ void k_bn_relu_bf16(const float* __restrict__ h,
                               const float* __restrict__ scale,
                               const float* __restrict__ shift,
                               const unsigned char* __restrict__ mask,
                               __bf16* __restrict__ out) {
    size_t i = (size_t)blockIdx.x * blockDim.x + threadIdx.x;
    if (i >= (size_t)NSITE * 64) return;
    const int    nidx = (int)(i & 63);
    const size_t s    = i >> 6;
    const float  mf   = mask[s] ? 1.0f : 0.0f;
    float v = (h[i] * scale[nidx] + shift[nidx]) * mf;
    out[i] = (__bf16)fmaxf(v, 0.0f);
}

// ---------------------------------------------------------------------------
// Final: residual 1x1 (WMMA, K=32) + BN2 apply + mask + fused 2x2x2 avg pool
// (float atomics into pre-zeroed output).
// ---------------------------------------------------------------------------
__global__ void k_final_res_pool(const __bf16* __restrict__ xb,      // [NSITE][32]
                                 const __bf16* __restrict__ wrfrag,  // prepacked [4][32][16]
                                 const float*  __restrict__ h2,      // [NSITE][64]
                                 const float*  __restrict__ scale,
                                 const float*  __restrict__ shift,
                                 const unsigned char* __restrict__ mask,
                                 float* __restrict__ outp) {         // [2,32,32,32,64]
    const int lane  = threadIdx.x & 31;
    const int wave  = threadIdx.x >> 5;
    const int mTile = wave >> 2;
    const int nTile = wave & 3;
    const int m     = lane & 15;
    const int khalf = lane >> 4;
    const int siteBase = blockIdx.x * 32 + mTile * 16;
    const int s = siteBase + m;
    const int n = (lane & 15) + nTile * 16;

    const __bf16* ap = xb + (size_t)s * CIN + khalf * 8;
    union { v16bf v; v8bf h[2]; } A;
    A.h[0] = *(const v8bf*)ap;
    A.h[1] = *(const v8bf*)(ap + 16);
    const v16bf Bv = ((const v16bf*)wrfrag)[nTile * 32 + lane];
    v8f acc = {};
    acc = __builtin_amdgcn_wmma_f32_16x16x32_bf16(
        false, A.v, false, Bv, (short)0, acc, false, false);

    const float sc = scale[n], sh = shift[n];
#pragma unroll
    for (int r = 0; r < 8; ++r) {
        const int mo = r + khalf * 8;
        const int so = siteBase + mo;
        const float mf = mask[so] ? 1.0f : 0.0f;
        float v = (h2[(size_t)so * 64 + n] * sc + sh) * mf + acc[r];
        const int wwx = so & 63, hhy = (so >> 6) & 63, ddz = (so >> 12) & 63, bbb = so >> 18;
        const size_t oi =
            ((((size_t)bbb * 32 + (ddz >> 1)) * 32 + (hhy >> 1)) * 32 + (wwx >> 1)) * 64 + n;
        atomicAdd(&outp[oi], v * 0.125f);
    }
}

// ---------------------------------------------------------------------------
// launch
// ---------------------------------------------------------------------------
extern "C" void kernel_launch(void* const* d_in, const int* in_sizes, int n_in,
                              void* d_out, int out_size, void* d_ws, size_t ws_size,
                              hipStream_t stream) {
    const float*         feats = (const float*)d_in[0];
    const unsigned char* mask  = (const unsigned char*)d_in[1];  // numpy bool_
    const float* w1  = (const float*)d_in[2];
    const float* b1  = (const float*)d_in[3];
    const float* g1  = (const float*)d_in[4];
    const float* be1 = (const float*)d_in[5];
    const float* w2  = (const float*)d_in[6];
    const float* b2  = (const float*)d_in[7];
    const float* g2  = (const float*)d_in[8];
    const float* be2 = (const float*)d_in[9];
    const float* wres = (const float*)d_in[10];
    float* out = (float*)d_out;
    (void)ws_size; (void)in_sizes; (void)n_in;

    // ---- workspace layout ----
    char*  ws    = (char*)d_ws;
    float* stats = (float*)ws;                 // 1024 floats
    size_t off   = 4096;
    __bf16* xb  = (__bf16*)(ws + off); off += (size_t)NSITE * CIN  * 2;   // 32 MB
    __bf16* h1b = (__bf16*)(ws + off); off += (size_t)NSITE * COUT * 2;   // 64 MB
    __bf16* f1  = (__bf16*)(ws + off); off += (size_t)27 * 1 * 2048 * 2;  // conv1 frags
    __bf16* f2  = (__bf16*)(ws + off); off += (size_t)27 * 2 * 2048 * 2;  // conv2 frags
    __bf16* fr  = (__bf16*)(ws + off); off += (size_t)1  * 1 * 2048 * 2;  // res frags
    off = (off + 255) & ~(size_t)255;
    float* hbuf = (float*)(ws + off);          // 128 MB, h1 then h2

    float* cnt    = stats + 0;
    float* sum1   = stats + 64,  *sq1 = stats + 128;
    float* scale1 = stats + 192, *shift1 = stats + 256;
    float* sum2   = stats + 320, *sq2 = stats + 384;
    float* scale2 = stats + 448, *shift2 = stats + 512;

    constexpr unsigned LDS1 = 27u * 1 * 2048 * 2;   // 110592 B
    constexpr unsigned LDS2 = 27u * 2 * 2048 * 2;   // 221184 B
    hipFuncSetAttribute(reinterpret_cast<const void*>(k_conv3<CIN>),
                        hipFuncAttributeMaxDynamicSharedMemorySize, LDS1);
    hipFuncSetAttribute(reinterpret_cast<const void*>(k_conv3<COUT>),
                        hipFuncAttributeMaxDynamicSharedMemorySize, LDS2);

    // 0) zero output (pool accumulates atomically) + stats
    k_zero_f32<<<(out_size + 255) / 256, 256, 0, stream>>>(out, out_size);
    k_zero_f32<<<4, 256, 0, stream>>>(stats, 1024);

    // 1) prepack weights into WMMA fragment layout (bf16)
    k_prepack<CIN,  27><<<(27 * 1 * 2048 + 255) / 256, 256, 0, stream>>>(w1,   f1);
    k_prepack<COUT, 27><<<(27 * 2 * 2048 + 255) / 256, 256, 0, stream>>>(w2,   f2);
    k_prepack<CIN,   1><<<(1  * 1 * 2048 + 255) / 256, 256, 0, stream>>>(wres, fr);

    // 2) x = feats * mask -> bf16, cnt = sum(mask)
    k_mask_convert<<<NSITE / 256, 256, 0, stream>>>(feats, mask, xb, cnt);

    // 3) conv1 (WMMA, TDM-staged weights) -> hbuf, BN1 stats
    k_conv3<CIN><<<NSITE / 32, 256, LDS1, stream>>>(xb, f1, b1, mask, hbuf, sum1, sq1);
    k_bn_finalize<<<1, 64, 0, stream>>>(cnt, sum1, sq1, g1, be1, scale1, shift1);

    // 4) BN1 + ReLU -> bf16
    k_bn_relu_bf16<<<(int)(((size_t)NSITE * 64) / 256), 256, 0, stream>>>(
        hbuf, scale1, shift1, mask, h1b);

    // 5) conv2 (WMMA) -> hbuf (reuse), BN2 stats
    k_conv3<COUT><<<NSITE / 32, 256, LDS2, stream>>>(h1b, f2, b2, mask, hbuf, sum2, sq2);
    k_bn_finalize<<<1, 64, 0, stream>>>(cnt, sum2, sq2, g2, be2, scale2, shift2);

    // 6) residual WMMA + BN2 apply + fused 2x2x2 avg pool
    k_final_res_pool<<<NSITE / 32, 256, 0, stream>>>(
        xb, fr, hbuf, scale2, shift2, mask, out);
}
